// TransformerDecoder_12987981103250
// MI455X (gfx1250) — compile-verified, matrix operand order checked by reading, not run on hardware
//
#include <hip/hip_runtime.h>

// ---------------------------------------------------------------------------
// Transformer decoder (B=256, E=512, H=8, DH=64, F=2048, V=10000, L=6, T=16)
// bf16 WMMA GEMMs (fp32 accumulate). All operands pre-packed in the WMMA
// register layout so the GEMM inner loop is pure global_load_b128 + v_wmma
// with zero control flow (inactive tail tiles run on clamped pointers and are
// discarded in the epilogue). Cross-attention over one memory token ==
// broadcast of V(memory): exact precompute crossadd[l][b,:].
// ---------------------------------------------------------------------------

#define BB 256
#define EE 512
#define HH 8
#define DHH 64
#define FF 2048
#define VV 10000
#define LL 6
#define TT 16

typedef __attribute__((ext_vector_type(16))) __bf16 v16bf;
typedef __attribute__((ext_vector_type(8)))  float  v8f;

__device__ __forceinline__ unsigned short f2bf_u(float f) {
  unsigned u = __builtin_bit_cast(unsigned, f);
  return (unsigned short)((u + 0x7FFFu + ((u >> 16) & 1u)) >> 16);
}

// K index for element e of the 16-bit WMMA A/B operand (lane-half hi).
__device__ __forceinline__ int kmap(int e, int hi) {
  return (e < 8 ? e : e + 8) + (hi ? 8 : 0);
}

// Inverse of kmap: given k in [0,32) -> (hi, e)
__device__ __forceinline__ void kinv(int k, int& hi, int& e) {
  hi = (k >> 3) & 1;
  e  = (k & 7) + (k >= 16 ? 8 : 0);
}

// Packed operand index for value (tileIdx, lane, e):
//   ((tileIdx)*32 + lane)*16 + e,   tileIdx = major*numKT + kt
__device__ __forceinline__ size_t pidx(int major, int numKT, int kt, int lane, int e) {
  return (((size_t)major * numKT + kt) * 32 + lane) * 16 + e;
}

// Scattered packed store of one scalar at logical (row, col) of a [*, Kdim] operand.
__device__ __forceinline__ void packed_store(unsigned short* __restrict__ dst, int numKT,
                                             int row, int col, float v) {
  int mt = row >> 4, m = row & 15;
  int kt = col >> 5, kl = col & 31;
  int hi, e; kinv(kl, hi, e);
  dst[pidx(mt, numKT, kt, m + 16 * hi, e)] = f2bf_u(v);
}

// ---------------------------------------------------------------------------
// Pack fp32 weight W (cols [c0, c0+N) of row-major [K, ld]) -> packed bf16 B-op.
// ---------------------------------------------------------------------------
__global__ void pack_w_kernel(const float* __restrict__ src, unsigned short* __restrict__ dst,
                              int K, int N, int c0, int ld) {
  int t = blockIdx.x * blockDim.x + threadIdx.x;
  int numKT = K >> 5;
  int total = numKT * (N >> 4) * 32;
  if (t >= total) return;
  int lane = t & 31;
  int tile = t >> 5;                 // tile = nt*numKT + kt
  int nt = tile / numKT;
  int kt = tile - nt * numKT;
  int col = c0 + nt * 16 + (lane & 15);
  int hi = lane >> 4;
  unsigned short* d = dst + ((size_t)tile * 32 + lane) * 16;
#pragma unroll
  for (int e = 0; e < 16; ++e)
    d[e] = f2bf_u(src[(size_t)((kt << 5) + kmap(e, hi)) * ld + col]);
}

// Pack fp32 activation A row-major [M, K] -> packed bf16 A-operand.
__global__ void pack_a_kernel(const float* __restrict__ src, unsigned short* __restrict__ dst,
                              int M, int K) {
  int t = blockIdx.x * blockDim.x + threadIdx.x;
  int numKT = K >> 5;
  int total = numKT * (M >> 4) * 32;
  if (t >= total) return;
  int lane = t & 31;
  int tile = t >> 5;                 // tile = mt*numKT + kt
  int mt = tile / numKT;
  int kt = tile - mt * numKT;
  int row = mt * 16 + (lane & 15);
  int hi = lane >> 4;
  unsigned short* d = dst + ((size_t)tile * 32 + lane) * 16;
#pragma unroll
  for (int e = 0; e < 16; ++e)
    d[e] = f2bf_u(src[(size_t)row * K + (kt << 5) + kmap(e, hi)]);
}

// ---------------------------------------------------------------------------
// GEMM: C[M,N] = act( Apk(bf16 packed) * Bpk(bf16 packed) + bias )
// Block: 128 threads = 4 waves. Wave tile 32x32 (2 A-tiles x 2 B-tiles,
// 4 accumulators -> 16 flop/B). Block tile 32(M) x 128(N). No LDS/barriers,
// and NO control flow in the K-loop: tail tiles (vocab GEMM only) use
// pointers clamped to tile 0 and are dropped in the epilogue.
// Outputs: Cf (fp32, nullable) and/or Cpk (packed bf16 A-op for next GEMM,
// nullable; requires N % 32 == 0).
// ---------------------------------------------------------------------------
__device__ __forceinline__ void epilogue_tile(const v8f& acc, int mt, int nt, int lane,
                                              const float* __restrict__ bias,
                                              float* __restrict__ Cf,
                                              unsigned short* __restrict__ Cpk,
                                              int ldc, int relu, int numKTout) {
  int hi = lane >> 4;
  int col = nt * 16 + (lane & 15);
  float bb = bias ? bias[col] : 0.f;
  int m0 = hi ? 8 : 0;
#pragma unroll
  for (int r = 0; r < 8; ++r) {
    float v = acc[r] + bb;
    if (relu) v = fmaxf(v, 0.f);
    int row = mt * 16 + m0 + r;
    if (Cf)  Cf[(size_t)row * ldc + col] = v;
    if (Cpk) packed_store(Cpk, numKTout, row, col, v);
  }
}

__global__ void __launch_bounds__(128)
gemm_bf16_wmma(const unsigned short* __restrict__ Apk, const unsigned short* __restrict__ Bpk,
               const float* __restrict__ bias, float* __restrict__ Cf,
               unsigned short* __restrict__ Cpk,
               int N, int K, int ldc, int relu) {
  const int wid  = threadIdx.x >> 5;
  const int lane = threadIdx.x & 31;
  const int numKT = K >> 5;
  const int mt0 = blockIdx.y * 2;
  const int ntA = blockIdx.x * 8 + wid * 2;
  const int ntB = ntA + 1;
  const bool actA = (ntA * 16) < N;
  const bool actB = (ntB * 16) < N;
  // Clamp inactive tail tiles (only the last block of the vocab GEMM) to tile 0:
  // loads stay valid, the dead accumulator is discarded in the epilogue.
  const int ntAe = actA ? ntA : 0;
  const int ntBe = actB ? ntB : 0;

  const uint4* a0p = (const uint4*)(Apk + pidx(mt0,     numKT, 0, lane, 0));
  const uint4* a1p = (const uint4*)(Apk + pidx(mt0 + 1, numKT, 0, lane, 0));
  const uint4* b0p = (const uint4*)(Bpk + pidx(ntAe,    numKT, 0, lane, 0));
  const uint4* b1p = (const uint4*)(Bpk + pidx(ntBe,    numKT, 0, lane, 0));

  v8f z = {0.f, 0.f, 0.f, 0.f, 0.f, 0.f, 0.f, 0.f};
  v8f acc00 = z, acc01 = z, acc10 = z, acc11 = z;

#pragma unroll 2
  for (int kt = 0; kt < numKT; ++kt) {
    union { uint4 u[2]; v16bf v; } a0, a1, b0, b1;
    a0.u[0] = a0p[0]; a0.u[1] = a0p[1];
    a1.u[0] = a1p[0]; a1.u[1] = a1p[1];
    b0.u[0] = b0p[0]; b0.u[1] = b0p[1];
    b1.u[0] = b1p[0]; b1.u[1] = b1p[1];
    a0p += 64; a1p += 64; b0p += 64; b1p += 64;   // 32 lanes * 16 bf16 = 64 uint4 / k-step
    acc00 = __builtin_amdgcn_wmma_f32_16x16x32_bf16(false, a0.v, false, b0.v, (short)0, acc00, false, false);
    acc10 = __builtin_amdgcn_wmma_f32_16x16x32_bf16(false, a1.v, false, b0.v, (short)0, acc10, false, false);
    acc01 = __builtin_amdgcn_wmma_f32_16x16x32_bf16(false, a0.v, false, b1.v, (short)0, acc01, false, false);
    acc11 = __builtin_amdgcn_wmma_f32_16x16x32_bf16(false, a1.v, false, b1.v, (short)0, acc11, false, false);
  }

  const int numKTout = N >> 5;
  if (actA) {
    epilogue_tile(acc00, mt0,     ntA, lane, bias, Cf, Cpk, ldc, relu, numKTout);
    epilogue_tile(acc10, mt0 + 1, ntA, lane, bias, Cf, Cpk, ldc, relu, numKTout);
  }
  if (actB) {
    epilogue_tile(acc01, mt0,     ntB, lane, bias, Cf, Cpk, ldc, relu, numKTout);
    epilogue_tile(acc11, mt0 + 1, ntB, lane, bias, Cf, Cpk, ldc, relu, numKTout);
  }
}

// ---------------------------------------------------------------------------
// Self-attention (no mask, S <= 16). One block per (b,h), blockDim = S*64.
// qkv: [B*S, 1536] fp32 (q|k|v). Output written as packed bf16 A-operand
// (rows = B*S tokens, Kdim = 512, numKT = 16) -- it only feeds the out-proj GEMM.
// ---------------------------------------------------------------------------
__global__ void self_attn_kernel(const float* __restrict__ qkv,
                                 unsigned short* __restrict__ outpk, int S) {
  int bh = blockIdx.x;
  int b = bh / HH, h = bh % HH;
  int t = threadIdx.x >> 6;   // query position
  int d = threadIdx.x & 63;   // head dim
  __shared__ float sc[16][16];
  __shared__ float pr[16][16];
  const float* base = qkv + (size_t)(b * S) * 1536;
  if (d < S) {
    const float* qr = base + (size_t)t * 1536 + h * DHH;
    const float* kr = base + (size_t)d * 1536 + EE + h * DHH;
    float s = 0.f;
#pragma unroll 8
    for (int i = 0; i < DHH; ++i) s += qr[i] * kr[i];
    sc[t][d] = s * 0.125f;   // 1/sqrt(64)
  }
  __syncthreads();
  if (d == 0) {
    float mx = -3.4e38f;
    for (int j = 0; j < S; ++j) mx = fmaxf(mx, sc[t][j]);
    float sum = 0.f;
    for (int j = 0; j < S; ++j) { float e = expf(sc[t][j] - mx); pr[t][j] = e; sum += e; }
    float inv = 1.f / sum;
    for (int j = 0; j < S; ++j) pr[t][j] *= inv;
  }
  __syncthreads();
  float o = 0.f;
  for (int j = 0; j < S; ++j)
    o += pr[t][j] * base[(size_t)j * 1536 + 2 * EE + h * DHH + d];
  packed_store(outpk, EE >> 5, b * S + t, h * DHH + d, o);
}

// ---------------------------------------------------------------------------
// out = LayerNorm(x + y)*g + b over E=512; also emits packed bf16 copy.
// yPerBatch: y is [B,512] broadcast over sequence. Safe in-place (out == x).
// ---------------------------------------------------------------------------
__global__ void __launch_bounds__(256)
ln_res_kernel(const float* __restrict__ x, const float* __restrict__ y, int yPerBatch, int S,
              const float* __restrict__ g, const float* __restrict__ be,
              float* __restrict__ out, unsigned short* __restrict__ outpk) {
  int row = blockIdx.x;
  int yrow = yPerBatch ? (row / S) : row;
  const float* xr = x + (size_t)row * EE;
  const float* yr = y + (size_t)yrow * EE;
  int tid = threadIdx.x;
  float v0 = xr[tid] + yr[tid];
  float v1 = xr[tid + 256] + yr[tid + 256];
  __shared__ float s1[256], s2[256];
  s1[tid] = v0 + v1;
  s2[tid] = v0 * v0 + v1 * v1;
  __syncthreads();
  for (int o = 128; o > 0; o >>= 1) {
    if (tid < o) { s1[tid] += s1[tid + o]; s2[tid] += s2[tid + o]; }
    __syncthreads();
  }
  float mean = s1[0] * (1.f / (float)EE);
  float var  = s2[0] * (1.f / (float)EE) - mean * mean;
  float inv  = rsqrtf(var + 1e-5f);
  float o0 = g[tid]       * (v0 - mean) * inv + be[tid];
  float o1 = g[tid + 256] * (v1 - mean) * inv + be[tid + 256];
  float* orow = out + (size_t)row * EE;
  orow[tid]       = o0;
  orow[tid + 256] = o1;
  packed_store(outpk, EE >> 5, row, tid,       o0);
  packed_store(outpk, EE >> 5, row, tid + 256, o1);
}

// Sinusoidal PE: pe[t, 2i] = sin(t * exp(2i * -ln(1e4)/E)), pe[t, 2i+1] = cos(.)
__global__ void pe_kernel(float* __restrict__ pe) {
  int t = blockIdx.x;       // 0..16
  int i = threadIdx.x;      // 0..255
  float ang = (float)t * expf((float)(2 * i) * (-9.210340371976184f / (float)EE));
  pe[t * EE + 2 * i]     = sinf(ang);
  pe[t * EE + 2 * i + 1] = cosf(ang);
}

__global__ void init_sos_kernel(const float* __restrict__ sos, float* __restrict__ inp) {
  int b = blockIdx.x, tid = threadIdx.x;
  float* dst = inp + (size_t)b * (TT + 1) * EE;
  dst[tid] = sos[tid];
  dst[tid + 256] = sos[tid + 256];
}

// x[b*S+t, :] = inp[b, t, :] + pe[t, :]   (fp32 + packed bf16)
__global__ void build_x_kernel(const float* __restrict__ inp, const float* __restrict__ pe,
                               float* __restrict__ x, unsigned short* __restrict__ xpk, int S) {
  int idx = blockIdx.x * 256 + threadIdx.x;
  int row = idx >> 9, e = idx & 511;
  int b = row / S, t = row - b * S;
  float v = inp[((size_t)b * (TT + 1) + t) * EE + e] + pe[t * EE + e];
  x[idx] = v;
  packed_store(xpk, EE >> 5, row, e, v);
}

// xlast[b,:] = x[b*S + S-1, :] as packed bf16 (feeds vocab GEMM only)
__global__ void xlast_kernel(const float* __restrict__ x, unsigned short* __restrict__ xlpk, int S) {
  int idx = blockIdx.x * 256 + threadIdx.x;   // over B*E
  int b = idx >> 9, e = idx & 511;
  packed_store(xlpk, EE >> 5, b, e, x[((size_t)b * S + (S - 1)) * EE + e]);
}

// Greedy argmax over V (first max wins); writes int sym + float sym into d_out.
__global__ void __launch_bounds__(256)
argmax_kernel(const float* __restrict__ logits, int ldl, int* __restrict__ sym,
              float* __restrict__ symout) {
  int b = blockIdx.x;
  const float* r = logits + (size_t)b * ldl;
  int tid = threadIdx.x;
  float best = -3.4e38f; int bi = 0;
  for (int v = tid; v < VV; v += 256) {
    float f = r[v];
    if (f > best) { best = f; bi = v; }
  }
  __shared__ float sv[256]; __shared__ int si[256];
  sv[tid] = best; si[tid] = bi;
  __syncthreads();
  for (int o = 128; o > 0; o >>= 1) {
    if (tid < o) {
      if (sv[tid + o] > sv[tid] || (sv[tid + o] == sv[tid] && si[tid + o] < si[tid])) {
        sv[tid] = sv[tid + o]; si[tid] = si[tid + o];
      }
    }
    __syncthreads();
  }
  if (tid == 0) { sym[b] = si[0]; symout[(size_t)b * TT] = (float)si[0]; }
}

// inp[b, snext, :] = sqrt(E) * embed_table[sym[b], :]
__global__ void emb_kernel(const float* __restrict__ table, const int* __restrict__ sym,
                           float* __restrict__ inp, int snext) {
  int b = blockIdx.x, tid = threadIdx.x;
  int s = sym[b];
  float* dst = inp + ((size_t)b * (TT + 1) + snext) * EE;
  const float* src = table + (size_t)s * EE;
  dst[tid]       = 22.62741699796952f * src[tid];
  dst[tid + 256] = 22.62741699796952f * src[tid + 256];
}

// ---------------------------------------------------------------------------
extern "C" void kernel_launch(void* const* d_in, const int* in_sizes, int n_in,
                              void* d_out, int out_size, void* d_ws, size_t ws_size,
                              hipStream_t stream) {
  (void)in_sizes; (void)n_in; (void)out_size; (void)ws_size;
  const float* latent   = (const float*)d_in[0];
  const float* embed    = (const float*)d_in[1];
  const float* sos      = (const float*)d_in[2];
  const float* proj1_W  = (const float*)d_in[3];
  const float* proj1_b  = (const float*)d_in[4];
  const float* proj2_W  = (const float*)d_in[5];
  const float* proj2_b  = (const float*)d_in[6];
  const float* qkv_w_s  = (const float*)d_in[7];
  const float* qkv_b_s  = (const float*)d_in[8];
  const float* out_w_s  = (const float*)d_in[9];
  const float* out_b_s  = (const float*)d_in[10];
  const float* qkv_w_c  = (const float*)d_in[11];
  const float* qkv_b_c  = (const float*)d_in[12];
  const float* out_w_c  = (const float*)d_in[13];
  const float* out_b_c  = (const float*)d_in[14];
  const float* ff_w1    = (const float*)d_in[15];
  const float* ff_b1    = (const float*)d_in[16];
  const float* ff_w2    = (const float*)d_in[17];
  const float* ff_b2    = (const float*)d_in[18];
  const float* ln1_g = (const float*)d_in[19];
  const float* ln1_b = (const float*)d_in[20];
  const float* ln2_g = (const float*)d_in[21];
  const float* ln2_b = (const float*)d_in[22];
  const float* ln3_g = (const float*)d_in[23];
  const float* ln3_b = (const float*)d_in[24];
  float* out_f = (float*)d_out;

  // ---- workspace partition (256B aligned) ----
  char* base = (char*)d_ws;
  size_t off = 0;
  auto alloc = [&](size_t bytes) -> char* {
    off = (off + 255) & ~(size_t)255;
    char* p = base + off;
    off += bytes;
    return p;
  };
  const size_t MMAX = (size_t)BB * TT;  // 4096 rows max
  // fp32 activations
  float* pe       = (float*)alloc(sizeof(float) * (TT + 1) * EE);
  float* inp      = (float*)alloc(sizeof(float) * BB * (TT + 1) * EE);
  float* x        = (float*)alloc(sizeof(float) * MMAX * EE);
  float* crossadd = (float*)alloc(sizeof(float) * LL * BB * EE);
  float* qkvbuf   = (float*)alloc(sizeof(float) * MMAX * 3 * EE);
  float* ybuf     = (float*)alloc(sizeof(float) * MMAX * EE);
  int*   symbuf   = (int*)alloc(sizeof(int) * BB);
  // packed bf16 activations (WMMA A-operand layout)
  unsigned short* xpk     = (unsigned short*)alloc(sizeof(short) * MMAX * EE);
  unsigned short* attnpk  = (unsigned short*)alloc(sizeof(short) * MMAX * EE);
  unsigned short* hpk     = (unsigned short*)alloc(sizeof(short) * MMAX * FF);
  unsigned short* latpk   = (unsigned short*)alloc(sizeof(short) * BB * EE);
  unsigned short* mempk   = (unsigned short*)alloc(sizeof(short) * BB * EE);
  unsigned short* vmempk  = (unsigned short*)alloc(sizeof(short) * BB * EE);
  unsigned short* xlastpk = (unsigned short*)alloc(sizeof(short) * BB * EE);
  // packed bf16 weights (WMMA B-operand layout)
  unsigned short* pWqkv_s = (unsigned short*)alloc(sizeof(short) * (size_t)LL * EE * 3 * EE);
  unsigned short* pWout_s = (unsigned short*)alloc(sizeof(short) * (size_t)LL * EE * EE);
  unsigned short* pWv_c   = (unsigned short*)alloc(sizeof(short) * (size_t)LL * EE * EE);
  unsigned short* pWout_c = (unsigned short*)alloc(sizeof(short) * (size_t)LL * EE * EE);
  unsigned short* pW1     = (unsigned short*)alloc(sizeof(short) * (size_t)LL * EE * FF);
  unsigned short* pW2     = (unsigned short*)alloc(sizeof(short) * (size_t)LL * FF * EE);
  unsigned short* pProj1  = (unsigned short*)alloc(sizeof(short) * (size_t)EE * EE);
  unsigned short* pProj2  = (unsigned short*)alloc(sizeof(short) * (size_t)EE * VV);

  auto pack = [&](const float* src, unsigned short* dst, int K, int N, int c0, int ld) {
    int total = (K >> 5) * (N >> 4) * 32;
    pack_w_kernel<<<(total + 127) / 128, 128, 0, stream>>>(src, dst, K, N, c0, ld);
  };
  auto gemm = [&](const unsigned short* Apk, const unsigned short* Bpk, const float* bias,
                  float* Cf, unsigned short* Cpk, int M, int N, int K, int ldc, int relu) {
    dim3 g((N + 127) / 128, M / 32);
    gemm_bf16_wmma<<<g, 128, 0, stream>>>(Apk, Bpk, bias, Cf, Cpk, N, K, ldc, relu);
  };

  // ---- one-time precompute (re-done every call: deterministic) ----
  pe_kernel<<<TT + 1, 256, 0, stream>>>(pe);
  init_sos_kernel<<<BB, 256, 0, stream>>>(sos, inp);
  {
    int total = (EE >> 5) * (BB >> 4) * 32;
    pack_a_kernel<<<(total + 127) / 128, 128, 0, stream>>>(latent, latpk, BB, EE);
  }
  pack(proj1_W, pProj1, EE, EE, 0, EE);
  pack(proj2_W, pProj2, EE, VV, 0, VV);
  for (int l = 0; l < LL; ++l) {
    pack(qkv_w_s + (size_t)l * EE * 3 * EE, pWqkv_s + (size_t)l * EE * 3 * EE, EE, 3 * EE, 0, 3 * EE);
    pack(out_w_s + (size_t)l * EE * EE,     pWout_s + (size_t)l * EE * EE,     EE, EE, 0, EE);
    pack(qkv_w_c + (size_t)l * EE * 3 * EE, pWv_c   + (size_t)l * EE * EE,     EE, EE, 2 * EE, 3 * EE);
    pack(out_w_c + (size_t)l * EE * EE,     pWout_c + (size_t)l * EE * EE,     EE, EE, 0, EE);
    pack(ff_w1   + (size_t)l * EE * FF,     pW1     + (size_t)l * EE * FF,     EE, FF, 0, FF);
    pack(ff_w2   + (size_t)l * FF * EE,     pW2     + (size_t)l * FF * EE,     FF, EE, 0, EE);
  }
  // memory = latent @ proj1_W + proj1_b (packed only: feeds GEMMs)
  gemm(latpk, pProj1, proj1_b, nullptr, mempk, BB, EE, EE, 0, 0);
  // crossadd[l] = (memory @ Wv_c + bv_c) @ Wout_c + bout_c   (softmax over 1 key == 1)
  for (int l = 0; l < LL; ++l) {
    gemm(mempk,  pWv_c   + (size_t)l * EE * EE, qkv_b_c + l * 3 * EE + 2 * EE,
         nullptr, vmempk, BB, EE, EE, 0, 0);
    gemm(vmempk, pWout_c + (size_t)l * EE * EE, out_b_c + l * EE,
         crossadd + (size_t)l * BB * EE, nullptr, BB, EE, EE, EE, 0);
  }

  // ---- autoregressive decode: full recompute each step (matches reference) ----
  for (int s = 0; s < TT; ++s) {
    int S = s + 1;
    int M = BB * S;
    build_x_kernel<<<M * 2, 256, 0, stream>>>(inp, pe, x, xpk, S);
    for (int l = 0; l < LL; ++l) {
      // self-attention sublayer
      gemm(xpk, pWqkv_s + (size_t)l * EE * 3 * EE, qkv_b_s + l * 3 * EE,
           qkvbuf, nullptr, M, 3 * EE, EE, 3 * EE, 0);
      self_attn_kernel<<<BB * HH, S * DHH, 0, stream>>>(qkvbuf, attnpk, S);
      gemm(attnpk, pWout_s + (size_t)l * EE * EE, out_b_s + l * EE,
           ybuf, nullptr, M, EE, EE, EE, 0);
      ln_res_kernel<<<M, 256, 0, stream>>>(x, ybuf, 0, S, ln1_g + l * EE, ln1_b + l * EE, x, xpk);
      // cross-attention sublayer (collapsed to per-batch constant)
      ln_res_kernel<<<M, 256, 0, stream>>>(x, crossadd + (size_t)l * BB * EE, 1, S,
                                           ln2_g + l * EE, ln2_b + l * EE, x, xpk);
      // feed-forward sublayer (hidden kept only as packed bf16)
      gemm(xpk, pW1 + (size_t)l * EE * FF, ff_b1 + l * FF, nullptr, hpk, M, FF, EE, 0, 1);
      gemm(hpk, pW2 + (size_t)l * FF * EE, ff_b2 + l * EE, ybuf, nullptr, M, EE, FF, EE, 0);
      ln_res_kernel<<<M, 256, 0, stream>>>(x, ybuf, 0, S, ln3_g + l * EE, ln3_b + l * EE, x, xpk);
    }
    xlast_kernel<<<(BB * EE) / 256, 256, 0, stream>>>(x, xlastpk, S);
    // logits for this step written straight into d_out (row stride 16*V)
    float* logits = out_f + (size_t)BB * TT + (size_t)s * VV;
    gemm(xlastpk, pProj2, proj2_b, logits, nullptr, BB, VV, EE, TT * VV, 0);
    argmax_kernel<<<BB, 256, 0, stream>>>(logits, TT * VV, symbuf, out_f + s);
    emb_kernel<<<BB, 256, 0, stream>>>(embed, symbuf, inp, s + 1);
  }
}
